// RGLRU_71390946394504
// MI455X (gfx1250) — matrix operand order repaired and checked by reading, not compiled, not run on hardware
//
#include <hip/hip_runtime.h>
#include <math.h>

// ---------------- problem constants ----------------
#define BB        4
#define SS        4096
#define D_IN      1024
#define D_HID     2048
#define D_INNER   1024
#define NBLK      16
#define MM        (BB * SS)          // 16384 rows
#define BC        (BB * D_INNER)     // 4096 scan channels
#define CL        128                // scan chunk length
#define NCHUNK    (SS / CL)          // 32 chunks

typedef __attribute__((ext_vector_type(16))) __bf16 v16bf;
typedef __attribute__((ext_vector_type(8)))  float  v8f;
typedef __attribute__((ext_vector_type(4)))  unsigned int v4u;
typedef __attribute__((ext_vector_type(8)))  int    v8i;
typedef __attribute__((ext_vector_type(4)))  int    v4i;

union FragU { v16bf v; uint4 q[2]; };

// =====================================================================
// TDM: async 2-D tile load (2-byte elements) global -> LDS.
// D# built per CDNA5 ISA §8.3/8.4: group0 {count=1, lds_addr, global_addr,
// type=2}, group1 {data_size=1 (2B), tensor_dim0/1, tile_dim0/1, stride0}.
// Groups 2/3 zero (<=2D tensor). Tracked by TENSORcnt.
// This toolchain exposes the 6-arg builtin form (extra int32x8 group).
// =====================================================================
static __device__ __forceinline__ void tdm_load_b16_2d(
    const void* gptr, unsigned lds_off,
    unsigned width, unsigned height, unsigned stride)   // in elements
{
    unsigned long long ga = (unsigned long long)(size_t)gptr;
    v4u g0;
    g0[0] = 1u;                                            // count=1, user mode
    g0[1] = lds_off;                                       // lds_addr (bytes)
    g0[2] = (unsigned)(ga & 0xFFFFFFFFu);                  // global_addr[31:0]
    g0[3] = (unsigned)((ga >> 32) & 0x01FFFFFFu)           // global_addr[56:32]
          | (2u << 30);                                    // type=2 (image)
    v8i g1 = {0, 0, 0, 0, 0, 0, 0, 0};
    g1[0] = (int)(1u << 16);                               // data_size=1 (2 bytes)
    g1[1] = (int)((width & 0xFFFFu) << 16);                // tensor_dim0[15:0]
    g1[2] = (int)((width >> 16) | ((height & 0xFFFFu) << 16));   // dim0 hi | dim1 lo
    g1[3] = (int)((height >> 16) | ((width & 0xFFFFu) << 16));   // dim1 hi | tile_dim0
    g1[4] = (int)(height & 0xFFFFu);                       // tile_dim1 (tile_dim2=0)
    g1[5] = (int)stride;                                   // tensor_dim0_stride[31:0]
    v4i z4 = {0, 0, 0, 0};
    v8i z8 = {0, 0, 0, 0, 0, 0, 0, 0};
    __builtin_amdgcn_tensor_load_to_lds(g0, g1, z4, z4, z8, 0);
}

// =====================================================================
// f32 -> bf16 pack, 8 elements/thread (b128 in, b128 out)
// =====================================================================
__global__ __launch_bounds__(256)
void convert_f32_bf16(const float* __restrict__ src, __bf16* __restrict__ dst, int n)
{
    const int i = (blockIdx.x * 256 + threadIdx.x) * 8;
    if (i >= n) return;
    const float4 a = *(const float4*)(src + i);
    const float4 b = *(const float4*)(src + i + 4);
    union { __bf16 h[8]; uint4 q; } u;
    u.h[0] = (__bf16)a.x; u.h[1] = (__bf16)a.y; u.h[2] = (__bf16)a.z; u.h[3] = (__bf16)a.w;
    u.h[4] = (__bf16)b.x; u.h[5] = (__bf16)b.y; u.h[6] = (__bf16)b.z; u.h[7] = (__bf16)b.w;
    *(uint4*)(dst + i) = u.q;
}

// =====================================================================
// Batched tiled transpose + convert: src [NB][K][N] f32 -> dst [NB][N][K] bf16
// 32x32 LDS tile (padded), 256 threads.
// =====================================================================
__global__ __launch_bounds__(256)
void transpose_convert(const float* __restrict__ src, __bf16* __restrict__ dst,
                       int K, int N)
{
    __shared__ float tile[32][33];
    const float* s = src + (size_t)blockIdx.z * K * N;
    __bf16*      d = dst + (size_t)blockIdx.z * K * N;
    const int nb = blockIdx.x * 32, kb = blockIdx.y * 32;
    const int tx = threadIdx.x & 31, ty = threadIdx.x >> 5;   // 32 x 8
#pragma unroll
    for (int r = 0; r < 32; r += 8)
        tile[ty + r][tx] = s[(size_t)(kb + ty + r) * N + (nb + tx)];
    __syncthreads();
#pragma unroll
    for (int r = 0; r < 32; r += 8)
        d[(size_t)(nb + ty + r) * K + (kb + tx)] = (__bf16)tile[tx][ty + r];
}

// =====================================================================
// bf16 WMMA GEMM:  C[M,N] = A[M,K](bf16) * Wt[N,K](bf16, pre-transposed) + bias
// Workgroup: 8 waves = 128 M-rows x one 16-col N slab.
// Wave 0 TDM-loads the 16xK B slab into LDS; all waves read B via ds_load_b128.
// A fragments: two global_load_b128 per K-step.
// =====================================================================
__global__ __launch_bounds__(256)
void gemm_bf16_wmma(const __bf16* __restrict__ A, const __bf16* __restrict__ Wt,
                    const float* __restrict__ bias, float* __restrict__ C,
                    __bf16* __restrict__ Cbf, int M, int N, int K)
{
    extern __shared__ __bf16 lds_b[];                 // [16][K]
    const int wave = threadIdx.x >> 5;
    const int l    = threadIdx.x & 31;
    const int half = l >> 4;
    const int lc   = l & 15;
    const int n0   = blockIdx.x * 16;
    const int m0   = blockIdx.y * 128 + wave * 16;
    const int col  = n0 + lc;

    if (wave == 0) {
        tdm_load_b16_2d(Wt + (size_t)n0 * K, (unsigned)(size_t)lds_b, K, 16, K);
        __builtin_amdgcn_s_wait_tensorcnt(0);
    }
    __syncthreads();

    const float bv = bias[col];
    v8f acc;
#pragma unroll
    for (int i = 0; i < 8; ++i) acc[i] = bv;

    const __bf16* arow = A + (size_t)(m0 + lc) * K;   // A row for this lane
    const __bf16* brow = lds_b + (size_t)lc * K;      // B column (= Wt row) in LDS

    for (int k0 = 0; k0 < K; k0 += 32) {
        FragU af, bf;
        // A: lane half 0 -> K {0..7,16..23}, half 1 -> {8..15,24..31}
        const __bf16* ap = arow + k0 + half * 8;
        af.q[0] = *(const uint4*)(ap);
        af.q[1] = *(const uint4*)(ap + 16);
        // B: lane half 0 -> K {0..7,16..23}, half 1 -> {8..15,24..31}
        const __bf16* bp = brow + k0 + half * 8;
        bf.q[0] = *(const uint4*)(bp);
        bf.q[1] = *(const uint4*)(bp + 16);
        acc = __builtin_amdgcn_wmma_f32_16x16x32_bf16(
                  false, af.v, false, bf.v, (short)0, acc, false, false);
    }
#pragma unroll
    for (int r = 0; r < 8; ++r) {
        const size_t idx = (size_t)(m0 + r + half * 8) * N + col;
        C[idx] = acc[r];
        if (Cbf) Cbf[idx] = (__bf16)acc[r];
    }
}

// =====================================================================
// Block-diagonal gate GEMMs (WMMA) fused with pointwise gate math.
// A fragments from xh_bf (b128), B fragments from pre-transposed bf16
// gate weights [NBLK][64][128] (contiguous b128). f32 xh used for the
// pointwise input values. Emits seq {a_re,a_im,x_re,x_im} in [t][chan].
// =====================================================================
__global__ __launch_bounds__(32)
void blockdiag_gates(const float*  __restrict__ xh,      // [M, D_HID] f32
                     const __bf16* __restrict__ xh_bf,   // [M, D_HID] bf16
                     const __bf16* __restrict__ WgxT,    // [NBLK,64,128]
                     const float*  __restrict__ bgx,
                     const __bf16* __restrict__ WgaT,
                     const float*  __restrict__ bga,
                     const float*  __restrict__ a_real_param,
                     const float*  __restrict__ a_imag_param,
                     float4* __restrict__ seq)            // [SS][BC]
{
    const int blk  = blockIdx.y;
    const int m0   = blockIdx.x * 16;
    const int l    = threadIdx.x;
    const int half = l >> 4;
    const int lc   = l & 15;

    v8f gx[4], ga[4];
#pragma unroll
    for (int nt = 0; nt < 4; ++nt) {
        const int d = blk * 64 + nt * 16 + lc;
        const float bx = bgx[d], ba = bga[d];
#pragma unroll
        for (int i = 0; i < 8; ++i) { gx[nt][i] = bx; ga[nt][i] = ba; }
    }

    const __bf16* xrow = xh_bf + (size_t)(m0 + lc) * D_HID + blk * 128;

#pragma unroll
    for (int kt = 0; kt < 4; ++kt) {                  // K = 128, 4 steps of 32
        const int k0 = kt * 32;
        FragU af;
        const __bf16* ap = xrow + k0 + half * 8;
        af.q[0] = *(const uint4*)(ap);
        af.q[1] = *(const uint4*)(ap + 16);
#pragma unroll
        for (int nt = 0; nt < 4; ++nt) {
            const int colb = nt * 16 + lc;            // 0..63 inside block
            FragU bx, ba;
            const __bf16* px = WgxT + ((size_t)blk * 64 + colb) * 128 + k0 + half * 8;
            const __bf16* pa = WgaT + ((size_t)blk * 64 + colb) * 128 + k0 + half * 8;
            bx.q[0] = *(const uint4*)(px); bx.q[1] = *(const uint4*)(px + 16);
            ba.q[0] = *(const uint4*)(pa); ba.q[1] = *(const uint4*)(pa + 16);
            gx[nt] = __builtin_amdgcn_wmma_f32_16x16x32_bf16(
                         false, af.v, false, bx.v, (short)0, gx[nt], false, false);
            ga[nt] = __builtin_amdgcn_wmma_f32_16x16x32_bf16(
                         false, af.v, false, ba.v, (short)0, ga[nt], false, false);
        }
    }

    const float PI = 3.14159265358979323846f;
#pragma unroll
    for (int nt = 0; nt < 4; ++nt) {
        const int d = blk * 64 + nt * 16 + lc;        // inner index 0..1023
        const float lr  = __logf(log1pf(__expf(a_real_param[d])));
        const float aim = a_imag_param[d];
#pragma unroll
        for (int r = 0; r < 8; ++r) {
            const int row = m0 + r + half * 8;        // global row = b*SS + s
            const float gxv = __frcp_rn(1.f + __expf(-gx[nt][r]));
            const float gav = __frcp_rn(1.f + __expf(-ga[nt][r]));
            const float magn  = __expf(gav * lr);
            const float theta = aim * gav * PI;
            float sn, cs; __sincosf(theta, &sn, &cs);
            const float norm = sqrtf(fmaxf(1.f - magn * magn, 1e-6f));
            const float xr = xh[(size_t)row * D_HID + d];
            const float xi = xh[(size_t)row * D_HID + D_INNER + d];
            const float g  = gxv * norm;
            const int b = row >> 12, s = row & (SS - 1);
            seq[(size_t)s * BC + (b * D_INNER + d)] =
                make_float4(magn * cs, magn * sn, g * xr, g * xi);
        }
    }
}

// =====================================================================
// 3-phase chunked complex linear scan:  h_t = a_t * h_{t-1} + x_t
// =====================================================================
__global__ __launch_bounds__(256)
void scan_phase1(const float4* __restrict__ seq, float4* __restrict__ carry)
{
    const int chan  = blockIdx.x * blockDim.x + threadIdx.x;
    const int chunk = blockIdx.y;
    float Ar = 1.f, Ai = 0.f, hr = 0.f, hi = 0.f;
    const int t0 = chunk * CL;
    for (int t = t0; t < t0 + CL; ++t) {
        __builtin_prefetch(&seq[(size_t)(t + 8) * BC + chan], 0, 1);
        const float4 v = seq[(size_t)t * BC + chan];
        const float nhr = fmaf(v.x, hr, fmaf(-v.y, hi, v.z));
        const float nhi = fmaf(v.x, hi, fmaf( v.y, hr, v.w));
        const float nAr = v.x * Ar - v.y * Ai;
        const float nAi = v.x * Ai + v.y * Ar;
        hr = nhr; hi = nhi; Ar = nAr; Ai = nAi;
    }
    carry[(size_t)chunk * BC + chan] = make_float4(Ar, Ai, hr, hi);
}

__global__ __launch_bounds__(256)
void scan_phase2(const float4* __restrict__ carry, float2* __restrict__ carryin)
{
    const int chan = blockIdx.x * blockDim.x + threadIdx.x;
    float hr = 0.f, hi = 0.f;
#pragma unroll
    for (int c = 0; c < NCHUNK; ++c) {
        carryin[(size_t)c * BC + chan] = make_float2(hr, hi);
        const float4 v = carry[(size_t)c * BC + chan];
        const float nhr = fmaf(v.x, hr, fmaf(-v.y, hi, v.z));
        const float nhi = fmaf(v.x, hi, fmaf( v.y, hr, v.w));
        hr = nhr; hi = nhi;
    }
}

__global__ __launch_bounds__(256)
void scan_phase3(const float4* __restrict__ seq, const float2* __restrict__ carryin,
                 __bf16* __restrict__ hm_bf,     // [M, D_HID] bf16 (real | imag)
                 float*  __restrict__ hlast)     // [BB, D_HID] f32
{
    const int chan  = blockIdx.x * blockDim.x + threadIdx.x;
    const int chunk = blockIdx.y;
    const float2 h0 = carryin[(size_t)chunk * BC + chan];
    float hr = h0.x, hi = h0.y;
    const int b = chan >> 10, d = chan & (D_INNER - 1);
    const int t0 = chunk * CL;
    for (int t = t0; t < t0 + CL; ++t) {
        __builtin_prefetch(&seq[(size_t)(t + 8) * BC + chan], 0, 1);
        const float4 v = seq[(size_t)t * BC + chan];
        const float nhr = fmaf(v.x, hr, fmaf(-v.y, hi, v.z));
        const float nhi = fmaf(v.x, hi, fmaf( v.y, hr, v.w));
        hr = nhr; hi = nhi;
        const size_t row = (size_t)b * SS + t;
        hm_bf[row * D_HID + d]           = (__bf16)hr;
        hm_bf[row * D_HID + D_INNER + d] = (__bf16)hi;
    }
    if (chunk == NCHUNK - 1) {
        hlast[b * D_HID + d]           = hr;
        hlast[b * D_HID + D_INNER + d] = hi;
    }
}

// =====================================================================
extern "C" void kernel_launch(void* const* d_in, const int* in_sizes, int n_in,
                              void* d_out, int out_size, void* d_ws, size_t ws_size,
                              hipStream_t stream)
{
    const float* x        = (const float*)d_in[0];
    const float* W_in     = (const float*)d_in[1];
    const float* b_in     = (const float*)d_in[2];
    const float* Wg_x     = (const float*)d_in[3];
    const float* bg_x     = (const float*)d_in[4];
    const float* Wg_a     = (const float*)d_in[5];
    const float* bg_a     = (const float*)d_in[6];
    const float* a_real_p = (const float*)d_in[7];
    const float* a_imag_p = (const float*)d_in[8];
    const float* W_out    = (const float*)d_in[9];
    const float* b_out    = (const float*)d_in[10];

    float* out   = (float*)d_out;                       // [B,S,D_IN]
    float* hlast = out + (size_t)MM * D_IN;             // [B,D_HID] tail

    // ---- workspace layout (all offsets 256B aligned) ----
    char* ws = (char*)d_ws;
    size_t off = 0;
    auto take = [&](size_t bytes) { char* p = ws + off; off += (bytes + 255) & ~(size_t)255; return p; };
    float*  xh      = (float*) take((size_t)MM * D_HID * 4);       // 128 MB
    float4* seq     = (float4*)take((size_t)SS * BC * 16);         // 256 MB
    float4* carry   = (float4*)take((size_t)NCHUNK * BC * 16);
    float2* carryin = (float2*)take((size_t)NCHUNK * BC * 8);
    __bf16* xbf     = (__bf16*)take((size_t)MM * D_IN * 2);        //  32 MB
    __bf16* xh_bf   = (__bf16*)take((size_t)MM * D_HID * 2);       //  64 MB
    __bf16* hm_bf   = (__bf16*)take((size_t)MM * D_HID * 2);       //  64 MB
    __bf16* WinT    = (__bf16*)take((size_t)D_IN * D_HID * 2);
    __bf16* WoutT   = (__bf16*)take((size_t)D_HID * D_IN * 2);
    __bf16* WgxT    = (__bf16*)take((size_t)NBLK * 128 * 64 * 2);
    __bf16* WgaT    = (__bf16*)take((size_t)NBLK * 128 * 64 * 2);

    // 0) bf16 conversions / weight transposes
    convert_f32_bf16<<<dim3((MM * D_IN) / (256 * 8)), 256, 0, stream>>>(x, xbf, MM * D_IN);
    transpose_convert<<<dim3(D_HID / 32, D_IN / 32, 1),  256, 0, stream>>>(W_in,  WinT,  D_IN,  D_HID);
    transpose_convert<<<dim3(D_IN / 32, D_HID / 32, 1),  256, 0, stream>>>(W_out, WoutT, D_HID, D_IN);
    transpose_convert<<<dim3(64 / 32, 128 / 32, NBLK),   256, 0, stream>>>(Wg_x,  WgxT,  128,   64);
    transpose_convert<<<dim3(64 / 32, 128 / 32, NBLK),   256, 0, stream>>>(Wg_a,  WgaT,  128,   64);

    // 1) xh = x @ W_in + b_in   (TDM-staged B, bf16 WMMA, fp32 acc; also bf16 copy)
    gemm_bf16_wmma<<<dim3(D_HID / 16, MM / 128), 256, (size_t)D_IN * 16 * 2, stream>>>(
        xbf, WinT, b_in, xh, xh_bf, MM, D_HID, D_IN);

    // 2) block-diagonal gates + pointwise -> scan sequence
    blockdiag_gates<<<dim3(MM / 16, NBLK), 32, 0, stream>>>(
        xh, xh_bf, WgxT, bg_x, WgaT, bg_a, a_real_p, a_imag_p, seq);

    // 3-5) chunked complex linear scan
    dim3 g3(BC / 256, NCHUNK);
    scan_phase1<<<g3, 256, 0, stream>>>(seq, carry);
    scan_phase2<<<dim3(BC / 256), 256, 0, stream>>>(carry, carryin);
    scan_phase3<<<g3, 256, 0, stream>>>(seq, carryin, hm_bf, hlast);

    // 6) out = h_merged @ W_out + b_out
    gemm_bf16_wmma<<<dim3(D_IN / 16, MM / 128), 256, (size_t)D_HID * 16 * 2, stream>>>(
        hm_bf, WoutT, b_out, out, (__bf16*)nullptr, MM, D_IN, D_HID);
}
